// AdaptiveCurvatureTargeting_34437047779991
// MI455X (gfx1250) — matrix operand order repaired and checked by reading, not compiled, not run on hardware
//
#include <hip/hip_runtime.h>
#include <hip/hip_bf16.h>

// ---------------- vector types for CDNA5 WMMA ----------------
typedef __attribute__((ext_vector_type(16))) _Float16 v16h;
typedef __attribute__((ext_vector_type(8)))  _Float16 v8h;
typedef __attribute__((ext_vector_type(8)))  float    v8f;

#define BB 4
#define TT 1024
#define PP 64
#define DD 128
#define NCLOUD (BB*TT)          // 4096
#define NPTS   (NCLOUD*PP)      // 262144

// D = A(16x32 f16) x B(32x16 f16) + C(16x16 f32)
__device__ __forceinline__ v8f wmma16(v16h a, v16h b, v8f c) {
    return __builtin_amdgcn_wmma_f32_16x16x32_f16(
        /*neg_a=*/false, a, /*neg_b=*/false, b,
        /*c_mod=*/(short)0, c, /*reuse_a=*/false, /*reuse_b=*/false);
}

// A-fragment, 16x32 tile of row-major f16 matrix (leading dim = ld elements).
// Layout: lane<16 -> row rowbase+lane, K = {kb..kb+15}
//         lane>=16 -> row rowbase+lane-16, K = {kb+16..kb+31}
__device__ __forceinline__ v16h frag_a(const _Float16* base, int ld,
                                       int rowbase, int kb, int lane) {
    int r  = rowbase + (lane & 15);
    int hi = (lane >> 4) << 4;            // 0 or 16
    const _Float16* p = base + r * ld + kb + hi;
    v8h lo  = *reinterpret_cast<const v8h*>(p);
    v8h hi8 = *reinterpret_cast<const v8h*>(p + 8);
    v16h out;
#pragma unroll
    for (int i = 0; i < 8; ++i) { out[i] = lo[i]; out[i + 8] = hi8[i]; }
    return out;
}

// B-fragment from a TRANSPOSED weight array Wt[n][k] (row-major, ld = K dim).
// Contiguous B k order used here (same as frag_a): VGPR j holds K = 2j,2j+1 (lanes 0-15, N=lane) and
// K = 16+2j,17+2j (lanes 16-31, N=lane-16)  ->  one contiguous 16-value load.
__device__ __forceinline__ v16h frag_bt(const _Float16* base, int ld,
                                        int nbase, int kb, int lane) {
    int n     = nbase + (lane & 15);
    int khalf = (lane >> 4) << 4;         // 0 or 16
    return *reinterpret_cast<const v16h*>(base + n * ld + kb + khalf);
}

// ---------------- kernel 1: weight convert + transpose to f16 ----------------
__global__ void prep_kernel(const float* __restrict__ cw1,
                            const float* __restrict__ cw2,
                            const float* __restrict__ hw1,
                            const float* __restrict__ hw2,
                            _Float16* __restrict__ w1t,    // [128][256]
                            _Float16* __restrict__ w2t,    // [64][128]
                            _Float16* __restrict__ hw1t,   // [32][160] zero-padded K
                            _Float16* __restrict__ hw2t) { // [16][32]
    int stride = gridDim.x * blockDim.x;
    int t = blockIdx.x * blockDim.x + threadIdx.x;
    for (int i = t; i < 128 * 256; i += stride) {
        int n = i >> 8, k = i & 255;
        w1t[i] = (_Float16)cw1[k * 128 + n];
    }
    for (int i = t; i < 64 * 128; i += stride) {
        int n = i >> 7, k = i & 127;
        w2t[i] = (_Float16)cw2[k * 64 + n];
    }
    for (int i = t; i < 32 * 160; i += stride) {
        int n = i / 160, k = i % 160;
        hw1t[i] = (_Float16)((k < 129) ? hw1[k * 32 + n] : 0.0f);
    }
    for (int i = t; i < 16 * 32; i += stride) {
        int n = i >> 5, k = i & 31;
        hw2t[i] = (_Float16)hw2[k * 16 + n];
    }
}

// ---------------- kernel 2: fully fused per-cloud pipeline ----------------
// one block (128 thr = 4 waves) per point cloud; positions read from HBM ONCE.
// Phases: stage -> sq -> Gram(WMMA)+topk -> ctx MLP (WMMA x80) -> hier MLP (WMMA x11)
//
// shared memory layout (bytes):
//   [0      .. 32768) in_s   f16 [64][256]  (pos | ctx)         -- live whole kernel
//   [32768  .. 36864) in2_s  f16 [64][32]   (tanh(depth)|zeros) -- live whole kernel
//   [36864  .. 53248) h1_s   f16 [64][128]  -- reused: g1_s f16[64][32] @36864,
//                                              g2_s f32[64][16] @40960
//   [53248  .. 69632) d2_s   f32 [64][64]   -- reused as h2_s f32[64][64]
//   [69632  .. 69888) sq_s   f32 [64]
#define SM_BYTES 69888
__global__ void fused_kernel(const float* __restrict__ pos,
                             const float* __restrict__ ctx,
                             const float* __restrict__ depth,
                             const _Float16* __restrict__ w1t,
                             const _Float16* __restrict__ w2t,
                             const _Float16* __restrict__ hw1t,
                             const _Float16* __restrict__ hw2t,
                             const float* __restrict__ cb1,
                             const float* __restrict__ cb2,
                             const float* __restrict__ cw3,
                             const float* __restrict__ cb3,
                             const float* __restrict__ hb1,
                             const float* __restrict__ hb2,
                             const float* __restrict__ hw3,
                             const float* __restrict__ hb3,
                             float* __restrict__ comb) {
    __shared__ __attribute__((aligned(32))) char smem[SM_BYTES];
    _Float16* in_s  = (_Float16*)(smem);
    _Float16* in2_s = (_Float16*)(smem + 32768);
    _Float16* h1_s  = (_Float16*)(smem + 36864);
    _Float16* g1_s  = (_Float16*)(smem + 36864);
    float*    g2_s  = (float*)(smem + 40960);
    float*    d2_s  = (float*)(smem + 53248);
    float*    h2_s  = (float*)(smem + 53248);
    float*    sq_s  = (float*)(smem + 69632);

    int cloud = blockIdx.x;
    int t = threadIdx.x, wave = t >> 5, lane = t & 31;
    const float* xp = pos   + (size_t)cloud * PP * DD;
    const float* cp = ctx   + (size_t)cloud * DD;
    const float* dp = depth + (size_t)cloud * PP;

    // ---- phase 1: stage inputs (single HBM pass over positions) ----
    {
        const float4* xp4 = reinterpret_cast<const float4*>(xp);
        for (int i = t; i < PP * DD / 4; i += 128) {
            float4 v = xp4[i];
            int g = i << 2;
            int p = g >> 7, k = g & 127;
            _Float16* dst = in_s + p * 256 + k;
            dst[0] = (_Float16)v.x; dst[1] = (_Float16)v.y;
            dst[2] = (_Float16)v.z; dst[3] = (_Float16)v.w;
        }
        const float4* cp4 = reinterpret_cast<const float4*>(cp);
        for (int i = t; i < PP * DD / 4; i += 128) {
            float4 v = cp4[i & 31];
            int g = i << 2;
            int p = g >> 7, k = g & 127;
            _Float16* dst = in_s + p * 256 + 128 + k;
            dst[0] = (_Float16)v.x; dst[1] = (_Float16)v.y;
            dst[2] = (_Float16)v.z; dst[3] = (_Float16)v.w;
        }
        for (int i = t; i < PP * 32; i += 128) {
            int p = i >> 5, k = i & 31;
            in2_s[i] = (_Float16)((k == 0) ? tanhf(dp[p]) : 0.0f);
        }
    }
    __syncthreads();

    // ---- phase 2: squared norms from LDS (no extra HBM traffic) ----
    if (t < PP) {
        float s = 0.0f;
        const _Float16* row = in_s + t * 256;
        for (int k = 0; k < DD; ++k) { float v = (float)row[k]; s += v * v; }
        sq_s[t] = s;
    }
    __syncthreads();

    int rowbase = wave * 16;
    int n = lane & 15, mh = (lane >> 4) << 3;

    // ---- phase 3: Gram = X X^T via WMMA (positions live in cols 0..127) ----
    for (int ct = 0; ct < 4; ++ct) {
        v8f acc = {};
#pragma unroll
        for (int kb = 0; kb < DD; kb += 32) {
            v16h a = frag_a(in_s, 256, rowbase, kb, lane);
            v16h b = frag_bt(in_s, 256, ct * 16, kb, lane); // B = X^T tile
            acc = wmma16(a, b, acc);
        }
#pragma unroll
        for (int i = 0; i < 8; ++i) {
            int r = rowbase + mh + i, c = ct * 16 + n;
            d2_s[r * PP + c] = sq_s[r] + sq_s[c] - 2.0f * acc[i];
        }
    }
    __syncthreads();

    // ---- phase 4: per-row 4 smallest distances -> local curvature ----
    float locv = 0.0f;
    if (t < PP) {
        float m0 = 1e30f, m1 = 1e30f, m2 = 1e30f, m3 = 1e30f;
        for (int c = 0; c < PP; ++c) {
            float dd = sqrtf(fmaxf(d2_s[t * PP + c], 1e-12f));
            if (dd < m3) {
                if (dd < m2) {
                    m3 = m2;
                    if (dd < m1) { m2 = m1; if (dd < m0) { m1 = m0; m0 = dd; } else m1 = dd; }
                    else m2 = dd;
                } else m3 = dd;
            }
        }
        float mean = (m1 + m2 + m3) * (1.0f / 3.0f);
        locv = -2.0f * tanhf(1.0f / (mean + 1e-6f));
    }
    __syncthreads();

    // ---- phase 5: ctx layer1 [64x256]@[256x128] + relu ----
    for (int nt = 0; nt < 8; ++nt) {
        float bias = cb1[nt * 16 + n];
        v8f acc;
#pragma unroll
        for (int i = 0; i < 8; ++i) acc[i] = bias;
#pragma unroll
        for (int kb = 0; kb < 256; kb += 32) {
            v16h a = frag_a(in_s, 256, rowbase, kb, lane);
            v16h b = frag_bt(w1t, 256, nt * 16, kb, lane);
            acc = wmma16(a, b, acc);
        }
#pragma unroll
        for (int i = 0; i < 8; ++i)
            h1_s[(rowbase + mh + i) * 128 + nt * 16 + n] = (_Float16)fmaxf(acc[i], 0.0f);
    }
    __syncthreads();

    // ---- phase 6: ctx layer2 [64x128]@[128x64] + relu (h2 reuses d2 space) ----
    for (int nt = 0; nt < 4; ++nt) {
        float bias = cb2[nt * 16 + n];
        v8f acc;
#pragma unroll
        for (int i = 0; i < 8; ++i) acc[i] = bias;
#pragma unroll
        for (int kb = 0; kb < 128; kb += 32) {
            v16h a = frag_a(h1_s, 128, rowbase, kb, lane);
            v16h b = frag_bt(w2t, 128, nt * 16, kb, lane);
            acc = wmma16(a, b, acc);
        }
#pragma unroll
        for (int i = 0; i < 8; ++i)
            h2_s[(rowbase + mh + i) * 64 + nt * 16 + n] = fmaxf(acc[i], 0.0f);
    }
    __syncthreads();

    // ---- phase 7a: ctx layer3 dot (register result) ----
    float ctxc = 0.0f;
    if (t < PP) {
        float s = cb3[0];
        for (int k = 0; k < 64; ++k) s += h2_s[t * 64 + k] * cw3[k];
        ctxc = s;
    }

    // ---- phase 7b: hier layer1 [64x160]@[160x32] + relu (g1 reuses h1 space)
    // K tiles 0..127 come from in_s (positions), tile 4 from in2_s (tanh|0) ----
    for (int nt = 0; nt < 2; ++nt) {
        float bias = hb1[nt * 16 + n];
        v8f acc;
#pragma unroll
        for (int i = 0; i < 8; ++i) acc[i] = bias;
#pragma unroll
        for (int kb = 0; kb < DD; kb += 32) {
            v16h a = frag_a(in_s, 256, rowbase, kb, lane);
            v16h b = frag_bt(hw1t, 160, nt * 16, kb, lane);
            acc = wmma16(a, b, acc);
        }
        {   // special K-tile 128..159
            v16h a = frag_a(in2_s, 32, rowbase, 0, lane);
            v16h b = frag_bt(hw1t, 160, nt * 16, 128, lane);
            acc = wmma16(a, b, acc);
        }
#pragma unroll
        for (int i = 0; i < 8; ++i)
            g1_s[(rowbase + mh + i) * 32 + nt * 16 + n] = (_Float16)fmaxf(acc[i], 0.0f);
    }
    __syncthreads();

    // ---- phase 8: hier layer2 [64x32]@[32x16] + tanh ----
    {
        float bias = hb2[n];
        v8f acc;
#pragma unroll
        for (int i = 0; i < 8; ++i) acc[i] = bias;
        v16h a = frag_a(g1_s, 32, rowbase, 0, lane);
        v16h b = frag_bt(hw2t, 32, 0, 0, lane);
        acc = wmma16(a, b, acc);
#pragma unroll
        for (int i = 0; i < 8; ++i)
            g2_s[(rowbase + mh + i) * 16 + n] = tanhf(acc[i]);
    }
    __syncthreads();

    // ---- phase 9: hier layer3 dot + final combine, single store ----
    if (t < PP) {
        float s = hb3[0];
        for (int k = 0; k < 16; ++k) s += g2_s[t * 16 + k] * hw3[k];
        comb[(size_t)cloud * PP + t] = 0.5f * locv + 0.3f * ctxc + 0.2f * s;
    }
}

// ---------------- kernel 3: per-p mean over (B,T) + momentum memory ----------------
__global__ void reduce_kernel(const float* __restrict__ comb,
                              const float* __restrict__ mem,
                              float* __restrict__ newmem) {
    __shared__ float part[256];
    int t = threadIdx.x;
    int p = t & 63, c = t >> 6;       // 4 chunks per p
    float s = 0.0f;
    for (int nIdx = c; nIdx < NCLOUD; nIdx += 4) s += comb[(size_t)nIdx * PP + p];
    part[t] = s;
    __syncthreads();
    if (t < PP) {
        float tot = part[t] + part[t + 64] + part[t + 128] + part[t + 192];
        float mean = tot * (1.0f / (float)NCLOUD);
        newmem[t] = 0.9f * mem[t] + 0.1f * mean;
    }
}

// ---------------- kernel 4: smoothing + clip ----------------
__global__ void final_kernel(const float* __restrict__ comb,
                             const float* __restrict__ newmem,
                             float* __restrict__ out) {
    int i = blockIdx.x * blockDim.x + threadIdx.x;
    if (i < NPTS) {
        float v = 0.9f * comb[i] + 0.1f * newmem[i & 63];
        out[i] = fminf(fmaxf(v, -5.0f), 2.0f);
    }
}

extern "C" void kernel_launch(void* const* d_in, const int* in_sizes, int n_in,
                              void* d_out, int out_size, void* d_ws, size_t ws_size,
                              hipStream_t stream) {
    (void)in_sizes; (void)n_in; (void)out_size; (void)ws_size;
    const float* positions = (const float*)d_in[0];
    const float* context   = (const float*)d_in[1];
    const float* depth     = (const float*)d_in[2];
    const float* cw1 = (const float*)d_in[3];
    const float* cb1 = (const float*)d_in[4];
    const float* cw2 = (const float*)d_in[5];
    const float* cb2 = (const float*)d_in[6];
    const float* cw3 = (const float*)d_in[7];
    const float* cb3 = (const float*)d_in[8];
    const float* hw1 = (const float*)d_in[9];
    const float* hb1 = (const float*)d_in[10];
    const float* hw2 = (const float*)d_in[11];
    const float* hb2 = (const float*)d_in[12];
    const float* hw3 = (const float*)d_in[13];
    const float* hb3 = (const float*)d_in[14];
    const float* curvmem = (const float*)d_in[15];
    float* out = (float*)d_out;

    // workspace layout
    char* ws = (char*)d_ws;
    float*    comb   = (float*)ws;                          // 262144 f32 (1 MB)
    float*    newmem = (float*)(ws + 1048576);              // 64 f32
    _Float16* w1t    = (_Float16*)(ws + 1048576 + 512);     // 128*256 f16
    _Float16* w2t    = (_Float16*)((char*)w1t + 128 * 256 * 2);
    _Float16* hw1t   = (_Float16*)((char*)w2t + 64 * 128 * 2);
    _Float16* hw2t   = (_Float16*)((char*)hw1t + 32 * 160 * 2);

    prep_kernel<<<64, 256, 0, stream>>>(cw1, cw2, hw1, hw2, w1t, w2t, hw1t, hw2t);
    fused_kernel<<<NCLOUD, 128, 0, stream>>>(positions, context, depth,
                                             w1t, w2t, hw1t, hw2t,
                                             cb1, cb2, cw3, cb3,
                                             hb1, hb2, hw3, hb3, comb);
    reduce_kernel<<<1, 256, 0, stream>>>(comb, curvmem, newmem);
    final_kernel<<<(NPTS + 255) / 256, 256, 0, stream>>>(comb, newmem, out);
}
